// TransformerBlock_61598420959545
// MI455X (gfx1250) — compile-verified, matrix operand order checked by reading, not run on hardware
//
#include <hip/hip_runtime.h>

// ---------------- problem constants ----------------
#define TOKEN 140
#define PDIM  4
#define SSEQ  8
#define DFF   1024
#define DKDIM 140
#define BTOT  8192
#define GB    4        // batch elements per workgroup
#define ROWS  16       // GB * PDIM  == one WMMA M-tile

typedef __bf16 bf16;
typedef __attribute__((ext_vector_type(16))) bf16  v16bf;
typedef __attribute__((ext_vector_type(8)))  bf16  v8bf;
typedef __attribute__((ext_vector_type(8)))  float v8f;
typedef __attribute__((ext_vector_type(4)))  unsigned int u32x4;
typedef __attribute__((ext_vector_type(8)))  unsigned int u32x8;

// ---------------- packed-weight layout in d_ws (bf16 elements) ----------------
// Each 32x16 (KxN) bf16 B-operand block is stored as 32 lanes * 16 bf16 = 512 elems,
// in the exact per-lane order V_WMMA_F32_16X16X32_BF16 expects.
static constexpr size_t BLK  = 512;
static constexpr size_t SZ_X = 9 * 5 * BLK;     // 140x140 padded -> 9 n-tiles, 5 k-tiles
static constexpr size_t SZ_H = 9 * 32 * BLK;    // 140x1024
static constexpr size_t SZ_W = 64 * 32 * BLK;   // 1024x1024
static constexpr size_t SZ_R = 64 * 5 * BLK;    // 1024x140
static constexpr size_t OFF_WQ  = 0;
static constexpr size_t OFF_WK  = OFF_WQ  + SZ_X;
static constexpr size_t OFF_WV  = OFF_WK  + SZ_X;
static constexpr size_t OFF_WCQ = OFF_WV  + SZ_X;
static constexpr size_t OFF_WCK = OFF_WCQ + SZ_H;
static constexpr size_t OFF_WCV = OFF_WCK + SZ_H;
static constexpr size_t OFF_WI  = OFF_WCV + SZ_H;
static constexpr size_t OFF_WF  = OFF_WI  + SZ_W;
static constexpr size_t OFF_WO  = OFF_WF  + SZ_W;
static constexpr size_t OFF_WZ  = OFF_WO  + SZ_W;
static constexpr size_t OFF_RI  = OFF_WZ  + SZ_W;
static constexpr size_t OFF_RF  = OFF_RI  + SZ_R;
static constexpr size_t OFF_RO  = OFF_RF  + SZ_R;
static constexpr size_t OFF_RZ  = OFF_RO  + SZ_R;

// ---------------- LDS layout (bytes) ----------------
static constexpr int SM_HBF   = 0;                    // bf16 h[2][16][1024]   65536
static constexpr int SM_C     = SM_HBF + 65536;       // f32  c[16][1024]      65536
static constexpr int SM_N     = SM_C   + 65536;       // f32  n[16][1024]      65536
static constexpr int SM_M     = SM_N   + 65536;       // f32  m[16][1024]      65536
static constexpr int SM_SI    = SM_M   + 65536;       // f32  si[16][140]       8960 (TDM tile, packed)
static constexpr int SM_SIB   = SM_SI  + 8960;        // bf16 si_bf[16][160]    5120
static constexpr int SM_Z2    = SM_SIB + 5120;        // bf16 z2[16][160]       5120
static constexpr int SM_QKV   = SM_Z2  + 5120;        // f32  qkv[3][16][144]  27648
static constexpr int SM_SC    = SM_QKV + 27648;       // f32  scores[16][4]      256
static constexpr int SM_AT    = SM_SC  + 256;         // f32  attn[16][4]        256
static constexpr int SM_MU    = SM_AT  + 256;         // f32  mu[16]              64
static constexpr int SM_RS    = SM_MU  + 64;          // f32  rstd[16]            64
static constexpr int SM_MK    = SM_RS  + 64;          // f32  mask[16]            64
static constexpr int SM_TOTAL = SM_MK  + 64;          // = 309696 bytes (< 320KB WGP LDS)

struct BiasPack {
    const float* b[14];   // wq,wk,wv,wcq,wck,wcv,wi,wf,wo,wz,ri,rf,ro,rz biases
    const float* ln_g;
    const float* ln_b;
};

// =====================================================================
// Prep: pack an f32 weight matrix W[dout][din] (used as x @ W^T) into
// bf16 B-operand blocks: block(nt,kt) holds B[k][n] = W[n][k] in the
// per-lane order of V_WMMA_F32_16X16X32_BF16's SRC1.
// =====================================================================
__global__ __launch_bounds__(256) void pack_w_kernel(
    const float* __restrict__ src, bf16* __restrict__ dst,
    int dout, int din, int ktcnt)
{
    const int blk = blockIdx.x;          // nt*ktcnt + kt
    const int nt  = blk / ktcnt;
    const int kt  = blk % ktcnt;
    bf16* d = dst + (size_t)blk * BLK;
    for (int e2 = threadIdx.x; e2 < (int)BLK; e2 += 256) {
        const int ln = e2 >> 4;          // lane 0..31
        const int e  = e2 & 15;          // element within lane's v16bf
        const int jj = e >> 1;           // dword index 0..7
        const int kk = (jj < 4 ? 2 * jj : 16 + 2 * (jj - 4)) + (e & 1) + ((ln >> 4) ? 8 : 0);
        const int k  = kt * 32 + kk;
        const int n  = nt * 16 + (ln & 15);
        float v = (n < dout && k < din) ? src[(size_t)n * din + k] : 0.0f;
        d[e2] = (bf16)v;
    }
}

// A operand (16x32 bf16) from a row-major bf16 LDS array.
__device__ __forceinline__ v16bf load_a(const bf16* base, int stride, int kbase, int lane)
{
    const int r  = lane & 15;
    const int hl = lane >> 4;
    const bf16* p  = base + r * stride + kbase + hl * 8;
    v8bf lo = *(const v8bf*)p;
    v8bf hi = *(const v8bf*)(p + 16);
    return __builtin_shufflevector(lo, hi, 0,1,2,3,4,5,6,7,8,9,10,11,12,13,14,15);
}

__device__ __forceinline__ v8f wmma_bf16(v16bf a, v16bf b, v8f c)
{
    return __builtin_amdgcn_wmma_f32_16x16x32_bf16(false, a, false, b, (short)0, c, false, false);
}

// =====================================================================
// Main persistent kernel: one workgroup = 4 batch elements = 16 rows,
// runs all S=8 recurrent steps with c/n/m/h state resident in LDS.
// =====================================================================
__global__ __launch_bounds__(256, 1) void xlstm_block_kernel(
    const float* __restrict__ state,   // [B,S,P,TOKEN]
    const float* __restrict__ maskg,   // [B,S,P,1]
    const bf16*  __restrict__ wpk,     // packed bf16 weights
    BiasPack pb,
    float* __restrict__ out_h,         // [B,P,DFF]
    float* __restrict__ out_attn)      // [B,S,1,P,P]
{
    extern __shared__ char smem[];
    bf16*  hb    = (bf16*) (smem + SM_HBF);
    float* c_s   = (float*)(smem + SM_C);
    float* n_s   = (float*)(smem + SM_N);
    float* m_s   = (float*)(smem + SM_M);
    float* si    = (float*)(smem + SM_SI);     // packed [16][140]
    bf16*  si_bf = (bf16*) (smem + SM_SIB);
    bf16*  z2_bf = (bf16*) (smem + SM_Z2);
    float* qkvf  = (float*)(smem + SM_QKV);
    float* sc_s  = (float*)(smem + SM_SC);
    float* at_s  = (float*)(smem + SM_AT);
    float* mu_s  = (float*)(smem + SM_MU);
    float* rs_s  = (float*)(smem + SM_RS);
    float* mk_s  = (float*)(smem + SM_MK);

    const int tid   = threadIdx.x;
    const int lane  = tid & 31;
    const int wave  = tid >> 5;
    const int bbase = blockIdx.x * GB;

    // zero recurrent state
    for (int i = tid; i < ROWS * DFF; i += 256) {
        c_s[i] = 0.0f; n_s[i] = 0.0f; m_s[i] = 0.0f;
        hb[i] = (bf16)0.0f; hb[ROWS * DFF + i] = (bf16)0.0f;
    }
    __syncthreads();

    int cur = 0;
    for (int s = 0; s < SSEQ; ++s) {
        bf16* hcur = hb + cur * ROWS * DFF;
        bf16* hnxt = hb + (cur ^ 1) * ROWS * DFF;

        // ---- stage 1: Tensor Data Mover pulls the s_i tile into LDS ----
        // 3D tile: dim0 = 140 f32 (contiguous), dim1 = P rows (stride 140),
        // dim2 = GB batches (stride S*P*TOKEN = 4480). TDM packs it
        // contiguously into si[16][140]. Issued once (wave 0), tracked by
        // TENSORcnt.
        if (wave == 0) {
            const unsigned long long ga =
                (unsigned long long)(const void*)(state + ((size_t)bbase * SSEQ + s) * PDIM * TOKEN);
            u32x4 g0;
            g0[0] = 1u;                                            // count=1, user descriptor
            g0[1] = (unsigned)SM_SI;                               // lds_addr (bytes)
            g0[2] = (unsigned)(ga & 0xFFFFFFFFu);                  // global_addr[31:0]
            g0[3] = (unsigned)((ga >> 32) & 0x01FFFFFFu) | (2u << 30); // global_addr[56:32] | type=2
            u32x8 g1;
            g1[0] = (2u << 16);                                    // data_size = 4 bytes
            g1[1] = ((unsigned)TOKEN << 16);                       // tensor_dim0 = 140 (lo bits)
            g1[2] = ((unsigned)PDIM << 16);                        // tensor_dim1 = 4 (lo bits)
            g1[3] = ((unsigned)TOKEN << 16);                       // tile_dim0 = 140
            g1[4] = (unsigned)PDIM | ((unsigned)GB << 16);         // tile_dim1 = 4, tile_dim2 = 4
            g1[5] = (unsigned)TOKEN;                               // tensor_dim0_stride = 140
            g1[6] = ((unsigned)(SSEQ * PDIM * TOKEN) << 16);       // tensor_dim1_stride = 4480 (lo)
            g1[7] = 0u;
            u32x4 g2; g2[0] = (unsigned)GB; g2[1] = 0u; g2[2] = 0u; g2[3] = 0u; // tensor_dim2 = 4
            u32x4 g3; g3[0] = 0u; g3[1] = 0u; g3[2] = 0u; g3[3] = 0u;
            asm volatile("tensor_load_to_lds %0, %1, %2, %3"
                         :: "s"(g0), "s"(g1), "s"(g2), "s"(g3)
                         : "memory");
            __builtin_amdgcn_s_wait_tensorcnt(0);
        }
        __syncthreads();

        // convert s_i -> padded bf16 A-operand; load mask
        for (int i = tid; i < ROWS * 160; i += 256) {
            const int r = i / 160, cc = i % 160;
            float v = (cc < TOKEN) ? si[r * TOKEN + cc] : 0.0f;
            si_bf[i] = (bf16)v;
        }
        if (tid < ROWS) {
            const int b = bbase + (tid >> 2), p = tid & 3;
            mk_s[tid] = maskg[((size_t)b * SSEQ + s) * PDIM + p];
        }
        __syncthreads();

        // ---- stage 2: q,k,v = si@Wx^T + h@Wc^T + biases (WMMA) ----
        for (int nt = wave; nt < 9; nt += 8) {
            for (int o = 0; o < 3; ++o) {
                v8f acc = {0.f,0.f,0.f,0.f,0.f,0.f,0.f,0.f};
                const bf16* wx = wpk + OFF_WQ + (size_t)o * SZ_X + (size_t)nt * (5 * BLK);
                #pragma unroll
                for (int kt = 0; kt < 5; ++kt) {
                    v16bf a = load_a(si_bf, 160, kt * 32, lane);
                    v16bf bfrag = *(const v16bf*)(wx + kt * BLK + lane * 16);
                    acc = wmma_bf16(a, bfrag, acc);
                }
                const bf16* wh = wpk + OFF_WCQ + (size_t)o * SZ_H + (size_t)nt * (32 * BLK);
                #pragma unroll 4
                for (int kt = 0; kt < 32; ++kt) {
                    __builtin_prefetch(wh + (kt + 4) * BLK + lane * 16, 0, 1); // speculative, branch-free
                    v16bf a = load_a(hcur, DFF, kt * 32, lane);
                    v16bf bfrag = *(const v16bf*)(wh + kt * BLK + lane * 16);
                    acc = wmma_bf16(a, bfrag, acc);
                }
                const int nl = lane & 15, hl = lane >> 4;
                const int n = nt * 16 + nl;
                const float bias = (n < DKDIM) ? (pb.b[o][n] + pb.b[3 + o][n]) : 0.0f;
                #pragma unroll
                for (int jj = 0; jj < 8; ++jj) {
                    const int r = jj + hl * 8;
                    qkvf[o * ROWS * 144 + r * 144 + n] = acc[jj] + bias;
                }
            }
        }
        __syncthreads();

        // ---- stage 3: scores + softmax (P=4 per batch) ----
        if (tid < 64) {
            const int pr = tid >> 2, qc = tid & 3;
            const int kr = (pr & ~3) + qc;
            const float4* qp = (const float4*)(qkvf + 0 * ROWS * 144 + pr * 144);
            const float4* kp = (const float4*)(qkvf + 1 * ROWS * 144 + kr * 144);
            float acc = 0.0f;
            #pragma unroll 5
            for (int d = 0; d < 35; ++d) {
                float4 a = qp[d], b = kp[d];
                acc += a.x * b.x + a.y * b.y + a.z * b.z + a.w * b.w;
            }
            sc_s[pr * 4 + qc] = acc * 0.084515425472851657f;  // 1/sqrt(140)
        }
        __syncthreads();
        if (tid < ROWS) {
            float s0 = sc_s[tid*4+0], s1 = sc_s[tid*4+1], s2 = sc_s[tid*4+2], s3 = sc_s[tid*4+3];
            float mx = fmaxf(fmaxf(s0, s1), fmaxf(s2, s3));
            float e0 = __expf(s0 - mx), e1 = __expf(s1 - mx), e2 = __expf(s2 - mx), e3 = __expf(s3 - mx);
            float inv = 1.0f / (e0 + e1 + e2 + e3);
            e0 *= inv; e1 *= inv; e2 *= inv; e3 *= inv;
            at_s[tid*4+0] = e0; at_s[tid*4+1] = e1; at_s[tid*4+2] = e2; at_s[tid*4+3] = e3;
            const int b = bbase + (tid >> 2), p = tid & 3;
            const size_t ob = (((size_t)b * SSEQ + s) * PDIM + p) * PDIM;
            out_attn[ob+0] = e0; out_attn[ob+1] = e1; out_attn[ob+2] = e2; out_attn[ob+3] = e3;
        }
        __syncthreads();

        // ---- stage 4: z1 = si + attn@v (in place into si) ----
        for (int i = tid; i < ROWS * TOKEN; i += 256) {
            const int r = i / TOKEN, cc = i % TOKEN;
            const int rb = r & ~3;
            float av = 0.0f;
            #pragma unroll
            for (int q2 = 0; q2 < 4; ++q2)
                av += at_s[r * 4 + q2] * qkvf[2 * ROWS * 144 + (rb + q2) * 144 + cc];
            si[r * TOKEN + cc] += av;
        }
        __syncthreads();

        // ---- stage 5: LayerNorm stats + z2 (bf16, padded to 160) ----
        if (tid < ROWS) {
            float mu = 0.0f, ssq = 0.0f;
            for (int d = 0; d < TOKEN; ++d) {
                float v = si[tid * TOKEN + d];
                mu += v; ssq += v * v;
            }
            mu /= (float)TOKEN;
            float var = ssq / (float)TOKEN - mu * mu;
            mu_s[tid] = mu;
            rs_s[tid] = rsqrtf(var + 1e-5f);
        }
        __syncthreads();
        for (int i = tid; i < ROWS * 160; i += 256) {
            const int r = i / 160, cc = i % 160;
            float v = 0.0f;
            if (cc < TOKEN)
                v = (si[r * TOKEN + cc] - mu_s[r]) * rs_s[r] * pb.ln_g[cc] + pb.ln_b[cc];
            z2_bf[i] = (bf16)v;
        }
        __syncthreads();

        // ---- stage 6: four gate GEMMs (WMMA) + fused xLSTM update ----
        for (int nt8 = 0; nt8 < 8; ++nt8) {
            const int nt = wave + nt8 * 8;
            v8f ai = {0.f,0.f,0.f,0.f,0.f,0.f,0.f,0.f};
            v8f af = ai, ao = ai, az = ai;
            const bf16* wip = wpk + OFF_WI + (size_t)nt * (32 * BLK);
            const bf16* wfp = wpk + OFF_WF + (size_t)nt * (32 * BLK);
            const bf16* wop = wpk + OFF_WO + (size_t)nt * (32 * BLK);
            const bf16* wzp = wpk + OFF_WZ + (size_t)nt * (32 * BLK);
            #pragma unroll 4
            for (int kt = 0; kt < 32; ++kt) {
                __builtin_prefetch(wip + (kt + 4) * BLK + lane * 16, 0, 1); // branch-free speculative
                __builtin_prefetch(wzp + (kt + 4) * BLK + lane * 16, 0, 1);
                v16bf a = load_a(hcur, DFF, kt * 32, lane);   // one A feeds 4 WMMAs
                ai = wmma_bf16(a, *(const v16bf*)(wip + kt * BLK + lane * 16), ai);
                af = wmma_bf16(a, *(const v16bf*)(wfp + kt * BLK + lane * 16), af);
                ao = wmma_bf16(a, *(const v16bf*)(wop + kt * BLK + lane * 16), ao);
                az = wmma_bf16(a, *(const v16bf*)(wzp + kt * BLK + lane * 16), az);
            }
            const bf16* rip = wpk + OFF_RI + (size_t)nt * (5 * BLK);
            const bf16* rfp = wpk + OFF_RF + (size_t)nt * (5 * BLK);
            const bf16* rop = wpk + OFF_RO + (size_t)nt * (5 * BLK);
            const bf16* rzp = wpk + OFF_RZ + (size_t)nt * (5 * BLK);
            #pragma unroll
            for (int kt = 0; kt < 5; ++kt) {
                v16bf a = load_a(z2_bf, 160, kt * 32, lane);
                ai = wmma_bf16(a, *(const v16bf*)(rip + kt * BLK + lane * 16), ai);
                af = wmma_bf16(a, *(const v16bf*)(rfp + kt * BLK + lane * 16), af);
                ao = wmma_bf16(a, *(const v16bf*)(rop + kt * BLK + lane * 16), ao);
                az = wmma_bf16(a, *(const v16bf*)(rzp + kt * BLK + lane * 16), az);
            }
            const int nl = lane & 15, hl = lane >> 4;
            const int col = nt * 16 + nl;
            const float bi = pb.b[6][col] + pb.b[10][col];
            const float bf = pb.b[7][col] + pb.b[11][col];
            const float bo = pb.b[8][col] + pb.b[12][col];
            const float bz = pb.b[9][col] + pb.b[13][col];
            #pragma unroll
            for (int jj = 0; jj < 8; ++jj) {
                const int r   = jj + hl * 8;
                const int idx = r * DFF + col;
                const float itil = ai[jj] + bi;
                const float ftil = af[jj] + bf;
                const float otil = ao[jj] + bo;
                const float ztil = az[jj] + bz;
                const float c0 = c_s[idx], n0 = n_s[idx], m0 = m_s[idx];
                const float hprev = (float)hcur[idx];
                const float mk = mk_s[r];
                const float mt = fmaxf(ftil + m0, itil);
                const float ie = __expf(itil - mt);
                const float fe = __expf(ftil + m0 - mt);
                const float og = 1.0f / (1.0f + __expf(-otil));
                const float nt_ = fe * n0 + ie;
                const float zz = tanhf(ztil);
                const float ct = (c0 * fe + zz * ie) * mk + (1.0f - mk) * c0;
                const float ht = og * (ct / (nt_ + 1e-8f)) * mk + (1.0f - mk) * hprev;
                c_s[idx] = ct; n_s[idx] = nt_; m_s[idx] = mt;
                hnxt[idx] = (bf16)ht;
            }
        }
        __syncthreads();
        cur ^= 1;
    }

    // ---- write final h (f32) ----
    const bf16* hfin = hb + cur * ROWS * DFF;
    for (int i = tid; i < ROWS * DFF; i += 256) {
        const int r = i / DFF, cc = i % DFF;
        const int b = bbase + (r >> 2), p = r & 3;
        out_h[((size_t)b * PDIM + p) * DFF + cc] = (float)hfin[i];
    }
}

// =====================================================================
// Host launcher
// =====================================================================
extern "C" void kernel_launch(void* const* d_in, const int* in_sizes, int n_in,
                              void* d_out, int out_size, void* d_ws, size_t ws_size,
                              hipStream_t stream) {
    (void)in_sizes; (void)n_in; (void)out_size; (void)ws_size;
    bf16* wpk = (bf16*)d_ws;

    struct PackDesc { int in_idx; size_t off; int dout, din, ntc, ktc; };
    const PackDesc pk[14] = {
        { 2, OFF_WQ,   140,  140,  9,  5},
        { 4, OFF_WK,   140,  140,  9,  5},
        { 6, OFF_WV,   140,  140,  9,  5},
        { 8, OFF_WCQ,  140, 1024,  9, 32},
        {10, OFF_WCK,  140, 1024,  9, 32},
        {12, OFF_WCV,  140, 1024,  9, 32},
        {14, OFF_WI,  1024, 1024, 64, 32},
        {16, OFF_WF,  1024, 1024, 64, 32},
        {18, OFF_WO,  1024, 1024, 64, 32},
        {20, OFF_WZ,  1024, 1024, 64, 32},
        {22, OFF_RI,  1024,  140, 64,  5},
        {24, OFF_RF,  1024,  140, 64,  5},
        {26, OFF_RO,  1024,  140, 64,  5},
        {28, OFF_RZ,  1024,  140, 64,  5},
    };
    for (int i = 0; i < 14; ++i) {
        pack_w_kernel<<<pk[i].ntc * pk[i].ktc, 256, 0, stream>>>(
            (const float*)d_in[pk[i].in_idx], wpk + pk[i].off,
            pk[i].dout, pk[i].din, pk[i].ktc);
    }

    BiasPack bp;
    for (int i = 0; i < 14; ++i) bp.b[i] = (const float*)d_in[3 + 2 * i];
    bp.ln_g = (const float*)d_in[30];
    bp.ln_b = (const float*)d_in[31];

    float* out_h    = (float*)d_out;
    float* out_attn = out_h + (size_t)BTOT * PDIM * DFF;

    xlstm_block_kernel<<<BTOT / GB, 256, SM_TOTAL, stream>>>(
        (const float*)d_in[0], (const float*)d_in[1], wpk, bp, out_h, out_attn);
}
